// GatedDeltaNet_59064390255185
// MI455X (gfx1250) — matrix operand order repaired and checked by reading, not compiled, not run on hardware
//
#include <hip/hip_runtime.h>
#include <hip/hip_bf16.h>

// ---------------- problem constants (match reference) ----------------
#define Bb 2
#define T 1024
#define D 2048
#define HK 8
#define HV 16
#define DK 128
#define DV 128
#define KW 4
#define KEY_DIM   (HK * DK)                 // 1024
#define VALUE_DIM (HV * DV)                 // 2048
#define CONV_DIM  (2 * KEY_DIM + VALUE_DIM) // 4096
#define MROWS (Bb * T)                      // 2048

// ---------------- WMMA types ----------------
typedef __bf16 bf16x16 __attribute__((ext_vector_type(16)));
typedef float  f32x8   __attribute__((ext_vector_type(8)));

union Frag { uint4 q[2]; bf16x16 v; };      // 32B per-lane WMMA operand

__device__ __forceinline__ unsigned short f2bf(float f) {
  unsigned int u = __float_as_uint(f);
  unsigned int r = u + 0x7FFFu + ((u >> 16) & 1u);   // round to nearest even
  return (unsigned short)(r >> 16);
}

// CDNA5 async global->LDS copy: per-lane 16B, tracked by ASYNCcnt.
__device__ __forceinline__ void async_load_b128(unsigned lds_off, const void* gptr) {
  asm volatile("global_load_async_to_lds_b128 %0, %1, off"
               :: "v"(lds_off), "v"((unsigned long long)(uintptr_t)gptr)
               : "memory");
}
__device__ __forceinline__ void wait_async0() {
  asm volatile("s_wait_asynccnt 0x0" ::: "memory");
}

// ---------------- fp32 -> bf16 conversion ----------------
__global__ void k_f2bf(const float* __restrict__ in, unsigned short* __restrict__ out, long n) {
  long i = (long)blockIdx.x * blockDim.x + threadIdx.x;
  if (i < n) out[i] = f2bf(in[i]);
}

// ---------------- fused fp32 -> bf16 + transpose: W[K,N] -> Wt[N,K] ----------------
__global__ __launch_bounds__(256) void k_f2bf_t(const float* __restrict__ in,
                                                unsigned short* __restrict__ out,
                                                int K, int N) {
  __shared__ unsigned short tile[32][33];
  int bx = blockIdx.x * 32;                 // N offset
  int by = blockIdx.y * 32;                 // K offset
  int tx = threadIdx.x & 31;
  int ty = threadIdx.x >> 5;                // 8 rows per pass
#pragma unroll
  for (int r = ty; r < 32; r += 8)
    tile[r][tx] = f2bf(in[(long)(by + r) * N + bx + tx]);
  __syncthreads();
#pragma unroll
  for (int r = ty; r < 32; r += 8)
    out[(long)(bx + r) * K + by + tx] = tile[tx][r];
}

// ---------------- bf16 WMMA GEMM: C[M,N] = A[M,K] * Bt[N,K]^T ----------------
// block = 256 threads (8 waves: 2 along M x 4 along N), tile 128x256, BK=32.
// LDS tiles stored in fragment-major order (per-lane 32B contiguous chunks),
// staged with async global->LDS b128 transfers, double-buffered.
#define BM 128
#define BN 256
#define BK 32

__global__ __launch_bounds__(256) void k_gemm_bf16(
    const unsigned short* __restrict__ A,    // [M,K] bf16 row-major
    const unsigned short* __restrict__ Bt,   // [N,K] bf16 row-major (pre-transposed)
    float* __restrict__ C, int M, int N, int K)
{
  __shared__ alignas(16) unsigned short AsF[2][8][32][16];   // [buf][sm][lane][e] 16KB
  __shared__ alignas(16) unsigned short BsF[2][16][32][16];  // [buf][sn][lane][e] 32KB

  const int tid  = threadIdx.x;
  const int lane = tid & 31;
  const int wid  = tid >> 5;
  const int wm   = wid & 1;     // 0..1 -> 64-row strip
  const int wn   = wid >> 1;    // 0..3 -> 64-col strip
  const int lr   = lane & 15;
  const int lh   = lane >> 4;
  const int bm   = blockIdx.y * BM;
  const int bn   = blockIdx.x * BN;

  f32x8 acc[4][4];
#pragma unroll
  for (int mi = 0; mi < 4; ++mi)
#pragma unroll
    for (int ni = 0; ni < 4; ++ni)
#pragma unroll
      for (int e = 0; e < 8; ++e) acc[mi][ni][e] = 0.0f;

  // Stage one 128x32 A tile + 256x32 B tile into fragment-major LDS via async DMA.
  auto stage = [&](int buf, int k0) {
    // A: 8 subtiles x 32 lanes x 2 halves = 512 x 16B -> 2 per thread
#pragma unroll
    for (int it = 0; it < 2; ++it) {
      int u  = tid + it * 256;
      int sm = u >> 6, l = (u >> 1) & 31, h = u & 1;
      int r  = sm * 16 + (l & 15);
      int kc = k0 + h * 16 + (l >> 4) * 8;          // ISA A-fragment K mapping
      async_load_b128((unsigned)(uintptr_t)&AsF[buf][sm][l][h * 8],
                      A + (long)(bm + r) * K + kc);
    }
    // B: 16 subtiles x 32 lanes x 2 halves = 1024 x 16B -> 4 per thread
#pragma unroll
    for (int it = 0; it < 4; ++it) {
      int u  = tid + it * 256;
      int sn = u >> 6, l = (u >> 1) & 31, h = u & 1;
      int c  = sn * 16 + (l & 15);
      int kc = k0 + h * 16 + (l >> 4) * 8;          // ISA B-fragment K mapping
      async_load_b128((unsigned)(uintptr_t)&BsF[buf][sn][l][h * 8],
                      Bt + (long)(bn + c) * K + kc);
    }
  };

  stage(0, 0);
  wait_async0();
  __syncthreads();

  const int nt = K / BK;
  for (int kt = 0; kt < nt; ++kt) {
    int p = kt & 1;
    if (kt + 1 < nt) stage(1 - p, (kt + 1) * BK);   // prefetch next tile (async)

    Frag af[4], bfm[4];
#pragma unroll
    for (int mi = 0; mi < 4; ++mi) {
      const uint4* s = (const uint4*)&AsF[p][wm * 4 + mi][lane][0];
      af[mi].q[0] = s[0];
      af[mi].q[1] = s[1];
    }
#pragma unroll
    for (int ni = 0; ni < 4; ++ni) {
      const uint4* s = (const uint4*)&BsF[p][wn * 4 + ni][lane][0];
      bfm[ni].q[0] = s[0];
      bfm[ni].q[1] = s[1];
    }
#pragma unroll
    for (int mi = 0; mi < 4; ++mi)
#pragma unroll
      for (int ni = 0; ni < 4; ++ni)
        acc[mi][ni] = __builtin_amdgcn_wmma_f32_16x16x32_bf16(
            false, af[mi].v, false, bfm[ni].v, (short)0, acc[mi][ni], false, false);

    if (kt + 1 < nt) {
      wait_async0();          // next buffer landed
      __syncthreads();        // one barrier per K-tile (double-buffered)
    }
  }

  // ---- store C (f32 C/D layout: VGPR i -> M = lh*8 + i, N = lr) ----
#pragma unroll
  for (int mi = 0; mi < 4; ++mi)
#pragma unroll
    for (int ni = 0; ni < 4; ++ni) {
      int col   = bn + wn * 64 + ni * 16 + lr;
      int rbase = bm + wm * 64 + mi * 16 + lh * 8;
#pragma unroll
      for (int i = 0; i < 8; ++i)
        C[(long)(rbase + i) * N + col] = acc[mi][ni][i];
    }
}

// ---------------- beta / g projections (N=16: memory-bound dot) ----------------
__global__ void k_beta_g(const float* __restrict__ hidden,
                         const float* __restrict__ Wb, const float* __restrict__ Wa,
                         const float* __restrict__ dt_bias, const float* __restrict__ A_log,
                         float* __restrict__ beta, float* __restrict__ gout)
{
  int i = blockIdx.x * blockDim.x + threadIdx.x;   // over MROWS*HV
  if (i >= MROWS * HV) return;
  int h  = i % HV;
  int bt = i / HV;
  const float* hp = hidden + (long)bt * D;
  float ab = 0.f, aa = 0.f;
  for (int d = 0; d < D; ++d) {
    float hv = hp[d];
    ab += hv * Wb[d * HV + h];
    aa += hv * Wa[d * HV + h];
  }
  beta[i] = 1.f / (1.f + __expf(-ab));
  float x  = aa + dt_bias[h];
  float sp = (x > 20.f) ? x : log1pf(__expf(x));   // softplus
  gout[i] = -__expf(A_log[h]) * sp;
}

// ---------------- depthwise causal conv (K=4) + SiLU ----------------
__global__ void k_conv_silu(const float* __restrict__ x, const float* __restrict__ w,
                            float* __restrict__ y)
{
  long i = (long)blockIdx.x * blockDim.x + threadIdx.x;
  if (i >= (long)MROWS * CONV_DIM) return;
  int c   = (int)(i % CONV_DIM);
  long bt = i / CONV_DIM;
  int t   = (int)(bt % T);
  long b  = bt / T;
  float acc = 0.f;
#pragma unroll
  for (int j = 0; j < KW; ++j) {
    int tt = t + j - (KW - 1);
    if (tt >= 0) acc += w[c * KW + j] * x[((long)b * T + tt) * CONV_DIM + c];
  }
  y[i] = acc / (1.f + __expf(-acc));               // SiLU
}

// ---------------- q/k L2 norm + GQA head expansion ----------------
__global__ __launch_bounds__(128) void k_qknorm(const float* __restrict__ mixed,
                                                float* __restrict__ qn, float* __restrict__ kn)
{
  int bth = blockIdx.x;            // over MROWS*HV
  int h   = bth % HV;
  int bt  = bth / HV;
  int hk  = h / (HV / HK);         // repeat_interleave source head
  int tid = threadIdx.x;           // 0..127 == DK
  float qv = mixed[(long)bt * CONV_DIM + hk * DK + tid];
  float kv = mixed[(long)bt * CONV_DIM + KEY_DIM + hk * DK + tid];
  __shared__ float rq[128], rk[128];
  rq[tid] = qv * qv;
  rk[tid] = kv * kv;
  __syncthreads();
  for (int s = 64; s > 0; s >>= 1) {
    if (tid < s) { rq[tid] += rq[tid + s]; rk[tid] += rk[tid + s]; }
    __syncthreads();
  }
  float qs = rsqrtf(rq[0] + 1e-6f) * 0.08838834764831845f;  // * DK^-0.5
  float ks = rsqrtf(rk[0] + 1e-6f);
  int ob = (bt * HV + h) * DK + tid;
  qn[ob] = qv * qs;
  kn[ob] = kv * ks;
}

// ---------------- gated delta-rule recurrence (serial over T) ----------------
// one block per (b,h); 256 threads; state S[128][128] f32 in registers:
// thread (g=tid>>7, dv=tid&127) owns S[g*64 .. g*64+63][dv].
__global__ __launch_bounds__(256) void k_recurrence(
    const float* __restrict__ qn, const float* __restrict__ kn,
    const float* __restrict__ mixed, const float* __restrict__ gArr,
    const float* __restrict__ betaArr, float* __restrict__ obuf)
{
  int bh = blockIdx.x;             // 0..Bb*HV-1
  int b  = bh / HV;
  int h  = bh % HV;
  int tid = threadIdx.x;
  int g   = tid >> 7;
  int dv  = tid & 127;

  float S[64];
#pragma unroll
  for (int j = 0; j < 64; ++j) S[j] = 0.f;

  __shared__ float sk[128], sq[128], sv[128];
  __shared__ float red[2][128];
  __shared__ float sgb[2];

  for (int t = 0; t < T; ++t) {
    long base = ((long)(b * T + t) * HV + h) * DK;
    if (tid < 128) {
      sk[tid] = kn[base + tid];
      sq[tid] = qn[base + tid];
    } else {
      int d2 = tid - 128;
      sv[d2] = mixed[(long)(b * T + t) * CONV_DIM + 2 * KEY_DIM + h * DV + d2];
    }
    if (tid == 0) {
      sgb[0] = gArr[(b * T + t) * HV + h];
      sgb[1] = betaArr[(b * T + t) * HV + h];
    }
    __syncthreads();

    float dec = __expf(sgb[0]);
    float btc = sgb[1];
    const float* kk = &sk[g * 64];
    const float* qq = &sq[g * 64];

    float p = 0.f;
#pragma unroll
    for (int j = 0; j < 64; ++j) { S[j] *= dec; p += kk[j] * S[j]; }
    red[g][dv] = p;
    __syncthreads();

    float kv   = red[0][dv] + red[1][dv];
    float vadj = (sv[dv] - kv) * btc;

    float p2 = 0.f;
#pragma unroll
    for (int j = 0; j < 64; ++j) { S[j] += kk[j] * vadj; p2 += qq[j] * S[j]; }
    __syncthreads();
    red[g][dv] = p2;
    __syncthreads();
    if (g == 0) obuf[base + dv] = red[0][dv] + red[1][dv];
    __syncthreads();
  }
}

// ---------------- gated RMSNorm * silu(z), write bf16 for final GEMM ----------------
__global__ __launch_bounds__(128) void k_normgate(const float* __restrict__ obuf,
                                                  const float* __restrict__ zbuf,
                                                  const float* __restrict__ norm_weight,
                                                  unsigned short* __restrict__ og)
{
  int bth = blockIdx.x;            // MROWS*HV
  int h   = bth % HV;
  int bt  = bth / HV;
  int tid = threadIdx.x;           // 0..127 == DV
  long idx = ((long)bt * HV + h) * DV + tid;
  float o = obuf[idx];
  __shared__ float r[128];
  r[tid] = o * o;
  __syncthreads();
  for (int s = 64; s > 0; s >>= 1) {
    if (tid < s) r[tid] += r[tid + s];
    __syncthreads();
  }
  float var = r[0] * (1.0f / DV);
  float on  = o * rsqrtf(var + 1e-6f) * norm_weight[tid];
  float z   = zbuf[(long)bt * VALUE_DIM + h * DV + tid];
  float sz  = z / (1.f + __expf(-z));
  og[(long)bt * VALUE_DIM + h * DV + tid] = f2bf(on * sz);
}

// ---------------- host-side launch ----------------
extern "C" void kernel_launch(void* const* d_in, const int* in_sizes, int n_in,
                              void* d_out, int out_size, void* d_ws, size_t ws_size,
                              hipStream_t stream) {
  const float* hidden      = (const float*)d_in[0];
  const float* W_qkv       = (const float*)d_in[1];
  const float* W_z         = (const float*)d_in[2];
  const float* W_b         = (const float*)d_in[3];
  const float* W_a         = (const float*)d_in[4];
  const float* conv_w      = (const float*)d_in[5];
  const float* dt_bias     = (const float*)d_in[6];
  const float* A_log       = (const float*)d_in[7];
  const float* norm_weight = (const float*)d_in[8];
  const float* W_out       = (const float*)d_in[9];
  float* out = (float*)d_out;
  (void)in_sizes; (void)n_in; (void)out_size; (void)ws_size;

  char* ws = (char*)d_ws;
  size_t off = 0;
  auto alloc = [&](size_t bytes) -> void* {
    void* p = ws + off;
    off = (off + bytes + 255) & ~(size_t)255;
    return p;
  };

  unsigned short* Abf     = (unsigned short*)alloc((size_t)MROWS * D * 2);
  unsigned short* WqkvT   = (unsigned short*)alloc((size_t)CONV_DIM * D * 2);   // [N,K]
  unsigned short* WzT     = (unsigned short*)alloc((size_t)VALUE_DIM * D * 2);  // [N,K]
  unsigned short* WoutT   = (unsigned short*)alloc((size_t)D * VALUE_DIM * 2);  // [N,K]
  float* mixed_pre        = (float*)alloc((size_t)MROWS * CONV_DIM * 4);
  float* mixed_post       = (float*)alloc((size_t)MROWS * CONV_DIM * 4);
  float* zbuf             = (float*)alloc((size_t)MROWS * VALUE_DIM * 4);
  float* betabuf          = (float*)alloc((size_t)MROWS * HV * 4);
  float* gbuf             = (float*)alloc((size_t)MROWS * HV * 4);
  float* qnbuf            = (float*)alloc((size_t)MROWS * HV * DK * 4);
  float* knbuf            = (float*)alloc((size_t)MROWS * HV * DK * 4);
  float* obuf             = (float*)alloc((size_t)MROWS * HV * DV * 4);
  unsigned short* ogbuf   = (unsigned short*)alloc((size_t)MROWS * VALUE_DIM * 2);

  // 1) conversions: activations straight, weights fused convert+transpose
  {
    long n = (long)MROWS * D;
    k_f2bf<<<dim3((n + 255) / 256), dim3(256), 0, stream>>>(hidden, Abf, n);
  }
  k_f2bf_t<<<dim3(CONV_DIM / 32, D / 32), dim3(256), 0, stream>>>(W_qkv, WqkvT, D, CONV_DIM);
  k_f2bf_t<<<dim3(VALUE_DIM / 32, D / 32), dim3(256), 0, stream>>>(W_z, WzT, D, VALUE_DIM);
  k_f2bf_t<<<dim3(D / 32, VALUE_DIM / 32), dim3(256), 0, stream>>>(W_out, WoutT, VALUE_DIM, D);

  // 2) mixed_pre = hidden @ W_qkv   [2048 x 4096]
  k_gemm_bf16<<<dim3(CONV_DIM / BN, MROWS / BM), dim3(256), 0, stream>>>(
      Abf, WqkvT, mixed_pre, MROWS, CONV_DIM, D);

  // 3) z = hidden @ W_z             [2048 x 2048]
  k_gemm_bf16<<<dim3(VALUE_DIM / BN, MROWS / BM), dim3(256), 0, stream>>>(
      Abf, WzT, zbuf, MROWS, VALUE_DIM, D);

  // 4) beta, g
  k_beta_g<<<dim3((MROWS * HV + 255) / 256), dim3(256), 0, stream>>>(
      hidden, W_b, W_a, dt_bias, A_log, betabuf, gbuf);

  // 5) causal depthwise conv + SiLU
  {
    long n = (long)MROWS * CONV_DIM;
    k_conv_silu<<<dim3((n + 255) / 256), dim3(256), 0, stream>>>(mixed_pre, conv_w, mixed_post);
  }

  // 6) q/k l2norm + GQA expansion
  k_qknorm<<<dim3(MROWS * HV), dim3(128), 0, stream>>>(mixed_post, qnbuf, knbuf);

  // 7) gated delta-rule recurrence
  k_recurrence<<<dim3(Bb * HV), dim3(256), 0, stream>>>(
      qnbuf, knbuf, mixed_post, gbuf, betabuf, obuf);

  // 8) gated RMSNorm * silu(z) -> bf16
  k_normgate<<<dim3(MROWS * HV), dim3(128), 0, stream>>>(obuf, zbuf, norm_weight, ogbuf);

  // 9) out = o_gated @ W_out        [2048 x 2048]
  k_gemm_bf16<<<dim3(D / BN, MROWS / BM), dim3(256), 0, stream>>>(
      ogbuf, WoutT, out, MROWS, D, VALUE_DIM);
}
